// ParallelDeblock_68521908241101
// MI455X (gfx1250) — matrix operand order repaired and checked by reading, not compiled, not run on hardware
//
#include <hip/hip_runtime.h>
#include <hip/hip_bf16.h>

typedef __attribute__((ext_vector_type(16))) __bf16 v16bf;
typedef __attribute__((ext_vector_type(8)))  float  v8f;

#define B_     16
#define V_IN_  1024
#define V_OUT_ 4096
#define C_IN_  128
#define C_OUT_ 256
#define K_     18
#define NNZ_   16384

// ---------------- workspace layout (bytes) ----------------
static constexpr size_t OFF_POOLF = 0;            // 16*4096*128 f32   = 33,554,432 B
static constexpr size_t OFF_POOLH = 33554432ULL;  // same in bf16      = 16,777,216 B
static constexpr size_t OFF_W1P   = 50331648ULL;  // 4*16  tiles * 1KB =     65,536 B
static constexpr size_t OFF_WFP   = 50397184ULL;  // 72*8  tiles       =    589,824 B
static constexpr size_t OFF_W2P   = 50987008ULL;  // 48*4  tiles       =    196,608 B
static constexpr size_t OFF_WDP   = 51183616ULL;  // 24*4  tiles       =     98,304 B
// total ~48.9 MiB

// ---------------- kernel 1: zero the fp32 pool ----------------
__global__ void k_zero_pool(float* __restrict__ poolF) {
    size_t i = (size_t)blockIdx.x * blockDim.x + threadIdx.x;   // in float4 units
    reinterpret_cast<float4*>(poolF)[i] = make_float4(0.f, 0.f, 0.f, 0.f);
}

// ---------------- kernel 2: sparse scatter-add pool ----------------
__global__ void k_pool_scatter(const float* __restrict__ x,
                               const float* __restrict__ vals,
                               const int*   __restrict__ rows,
                               const int*   __restrict__ cols,
                               float* __restrict__ poolF) {
    int i = blockIdx.x;          // nnz index
    int c = threadIdx.x;         // channel (128 threads)
    int r  = rows[i];
    int cl = cols[i];
    float v = vals[i];
    const float* xp = x     + (size_t)cl * C_IN_ + c;
    float*       op = poolF + (size_t)r  * C_IN_ + c;
#pragma unroll
    for (int b = 0; b < B_; ++b) {
        float g = xp[(size_t)b * V_IN_ * C_IN_] * v;
        __hip_atomic_fetch_add(op + (size_t)b * V_OUT_ * C_IN_, g,
                               __ATOMIC_RELAXED, __HIP_MEMORY_SCOPE_AGENT);
    }
}

// ---------------- kernel 3: pool fp32 -> bf16 ----------------
__global__ void k_cvt_pool(const float* __restrict__ poolF, __bf16* __restrict__ poolH) {
    size_t i = ((size_t)blockIdx.x * blockDim.x + threadIdx.x) * 4;
    float4 f = *reinterpret_cast<const float4*>(poolF + i);
    union { __bf16 h[4]; uint2 u; } r;
    r.h[0] = (__bf16)f.x; r.h[1] = (__bf16)f.y;
    r.h[2] = (__bf16)f.z; r.h[3] = (__bf16)f.w;
    *reinterpret_cast<uint2*>(poolH + i) = r.u;
}

// ---------------- kernel 4: pack weights into WMMA-B tile layout ----------------
// Tile = 32(K) x 16(N) bf16, stored as 32 lanes * 16 halves, lane-major.
// Lane l: N = l&15, Klocal = (l>=16 ? 8 : 0) + (i&7) + (i>=8 ? 16 : 0)  (i = element 0..15).
__global__ void k_pack_weights(const float* __restrict__ W1, const float* __restrict__ Wf,
                               const float* __restrict__ W2, const float* __restrict__ Wd,
                               __bf16* __restrict__ W1p, __bf16* __restrict__ Wfp,
                               __bf16* __restrict__ W2p, __bf16* __restrict__ Wdp) {
    int t = blockIdx.x;
    const float* W; __bf16* dst; int Nw, ntiles, tt;
    if (t < 64)       { W = W1; dst = W1p; Nw = 256; ntiles = 16; tt = t;       }
    else if (t < 640) { W = Wf; dst = Wfp; Nw = 128; ntiles = 8;  tt = t - 64;  }
    else if (t < 832) { W = W2; dst = W2p; Nw = 64;  ntiles = 4;  tt = t - 640; }
    else              { W = Wd; dst = Wdp; Nw = 64;  ntiles = 4;  tt = t - 832; }
    int kt = tt / ntiles;
    int nt = tt % ntiles;
#pragma unroll
    for (int q = 0; q < 2; ++q) {
        int e    = threadIdx.x * 2 + q;       // 0..511 within tile
        int lane = e >> 4;
        int i    = e & 15;
        int Kl   = ((lane & 16) ? 8 : 0) + (i & 7) + ((i & 8) ? 16 : 0);
        int N    = lane & 15;
        dst[(size_t)tt * 512 + e] =
            (__bf16)W[(size_t)(kt * 32 + Kl) * Nw + nt * 16 + N];
    }
}

// ---------------- WMMA helpers ----------------
__device__ __forceinline__ v16bf loadA16(const __bf16* p) {
    // 16-bit A 16x32 layout: per lane two contiguous 16B chunks (K base..+7, K base+16..+23)
    union { float4 q[2]; v16bf v; } u;
    u.q[0] = *reinterpret_cast<const float4*>(p);
    u.q[1] = *reinterpret_cast<const float4*>(p + 16);
    return u.v;
}
__device__ __forceinline__ v16bf loadB16(const __bf16* p) {
    union { float4 q[2]; v16bf v; } u;
    u.q[0] = *reinterpret_cast<const float4*>(p);
    u.q[1] = *reinterpret_cast<const float4*>(p + 8);
    return u.v;
}
__device__ __forceinline__ v8f wmma_bf16(v16bf a, v16bf b, v8f c) {
    return __builtin_amdgcn_wmma_f32_16x16x32_bf16(false, a, false, b, (short)0, c,
                                                   false, false);
}

// ---------------- kernel 5: fused spiral-gather GEMM + bias + ReLU ----------------
// grid = V_OUT/2 workgroups of 256 threads (8 wave32 waves).
// Each workgroup handles TWO vertices (two 16-row M-tiles = 16 batches each),
// so every B tile load feeds 2 WMMAs. Wave w owns 32 output columns of both
// short_cut and f (waves 0-3: p/Wf 18 nbrs, 4-5: p_2d3 12 nbrs, 6-7: p_d3 6 nbrs).
__global__ void __launch_bounds__(256)
k_spiral_gemm(const __bf16* __restrict__ poolH,
              const __bf16* __restrict__ W1p, const __bf16* __restrict__ Wfp,
              const __bf16* __restrict__ W2p, const __bf16* __restrict__ Wdp,
              const float*  __restrict__ b1,  const float*  __restrict__ bd3,
              const float*  __restrict__ b2d3,const float*  __restrict__ bfb,
              const int*    __restrict__ spiral,
              float* __restrict__ out) {
    const int v0   = blockIdx.x * 2;
    const int v1   = v0 + 1;
    const int lane = threadIdx.x & 31;
    const int wave = threadIdx.x >> 5;

    // per-wave role
    const __bf16* fW; const float* fB; int nN, fNtiles, nt0, outColBase;
    if (wave < 4)      { fW = Wfp; fB = bfb;  nN = 18; fNtiles = 8; nt0 = 2 * wave;       outColBase = 32 * wave; }
    else if (wave < 6) { fW = W2p; fB = b2d3; nN = 12; fNtiles = 4; nt0 = 2 * (wave - 4); outColBase = 128 + 32 * (wave - 4); }
    else               { fW = Wdp; fB = bd3;  nN = 6;  fNtiles = 4; nt0 = 2 * (wave - 6); outColBase = 192 + 32 * (wave - 6); }
    const int ntS0 = outColBase >> 4;   // == 2*wave; W1 has 16 N-tiles over 256 cols

    // accumulators: S=short_cut, F=f ; a/b = vertex v0/v1 ; 0/1 = N-tile
    v8f accS0a = {}, accS1a = {}, accF0a = {}, accF1a = {};
    v8f accS0b = {}, accS1b = {}, accF0b = {}, accF1b = {};

    const int aM    = lane & 15;                 // batch index (row of M-tile)
    const int aKoff = (lane & 16) ? 8 : 0;       // K half-select within 32-chunk
    const __bf16* aRowBase = poolH + (size_t)aM * V_OUT_ * C_IN_;

    // ---- neighbor 0 (feeds short_cut AND f) ----
    {
        const int u0 = spiral[v0 * K_ + 0];
        const int u1 = spiral[v1 * K_ + 0];
        const __bf16* ab0 = aRowBase + (size_t)u0 * C_IN_ + aKoff;
        const __bf16* ab1 = aRowBase + (size_t)u1 * C_IN_ + aKoff;
#pragma unroll
        for (int s = 0; s < 4; ++s) {
            v16bf a0 = loadA16(ab0 + s * 32);
            v16bf a1 = loadA16(ab1 + s * 32);
            v16bf bf0 = loadB16(fW + ((size_t)(s * fNtiles + nt0)) * 512 + lane * 16);
            v16bf bf1 = loadB16(fW + ((size_t)(s * fNtiles + nt0 + 1)) * 512 + lane * 16);
            accF0a = wmma_bf16(a0, bf0, accF0a);
            accF0b = wmma_bf16(a1, bf0, accF0b);
            accF1a = wmma_bf16(a0, bf1, accF1a);
            accF1b = wmma_bf16(a1, bf1, accF1b);
            v16bf bs0 = loadB16(W1p + ((size_t)(s * 16 + ntS0)) * 512 + lane * 16);
            v16bf bs1 = loadB16(W1p + ((size_t)(s * 16 + ntS0 + 1)) * 512 + lane * 16);
            accS0a = wmma_bf16(a0, bs0, accS0a);
            accS0b = wmma_bf16(a1, bs0, accS0b);
            accS1a = wmma_bf16(a0, bs1, accS1a);
            accS1b = wmma_bf16(a1, bs1, accS1b);
        }
    }
    // ---- neighbors 1..nN-1 (f only) ----
    for (int n = 1; n < nN; ++n) {
        const int u0 = spiral[v0 * K_ + n];
        const int u1 = spiral[v1 * K_ + n];
        const __bf16* ab0 = aRowBase + (size_t)u0 * C_IN_ + aKoff;
        const __bf16* ab1 = aRowBase + (size_t)u1 * C_IN_ + aKoff;
#pragma unroll
        for (int s = 0; s < 4; ++s) {
            v16bf a0 = loadA16(ab0 + s * 32);
            v16bf a1 = loadA16(ab1 + s * 32);
            const int kt = n * 4 + s;
            v16bf bf0 = loadB16(fW + ((size_t)(kt * fNtiles + nt0)) * 512 + lane * 16);
            v16bf bf1 = loadB16(fW + ((size_t)(kt * fNtiles + nt0 + 1)) * 512 + lane * 16);
            accF0a = wmma_bf16(a0, bf0, accF0a);
            accF0b = wmma_bf16(a1, bf0, accF0b);
            accF1a = wmma_bf16(a0, bf1, accF1a);
            accF1b = wmma_bf16(a1, bf1, accF1b);
        }
    }

    // ---- epilogue: out = relu(sc + f + b1 + bias_f); C/D layout:
    // VGPR g, lane l: M = g + (l>=16 ? 8 : 0), N = l&15
    const int N  = lane & 15;
    const int mo = (lane & 16) ? 8 : 0;
    const int localCol = (wave < 4) ? (32 * wave)
                       : (wave < 6) ? (32 * (wave - 4))
                                    : (32 * (wave - 6));
    const int j0 = outColBase + N;
    const int j1 = j0 + 16;
    const float bias0 = b1[j0] + fB[localCol + N];
    const float bias1 = b1[j1] + fB[localCol + 16 + N];
#pragma unroll
    for (int g = 0; g < 8; ++g) {
        const int bm = g + mo;   // batch
        float* oa = out + (((size_t)bm * V_OUT_ + v0) * C_OUT_);
        float* ob = out + (((size_t)bm * V_OUT_ + v1) * C_OUT_);
        oa[j0] = fmaxf(accS0a[g] + accF0a[g] + bias0, 0.f);
        oa[j1] = fmaxf(accS1a[g] + accF1a[g] + bias1, 0.f);
        ob[j0] = fmaxf(accS0b[g] + accF0b[g] + bias0, 0.f);
        ob[j1] = fmaxf(accS1b[g] + accF1b[g] + bias1, 0.f);
    }
}

// ---------------- launcher ----------------
extern "C" void kernel_launch(void* const* d_in, const int* in_sizes, int n_in,
                              void* d_out, int out_size, void* d_ws, size_t ws_size,
                              hipStream_t stream) {
    const float* x     = (const float*)d_in[0];
    const float* vals  = (const float*)d_in[1];
    const float* W1    = (const float*)d_in[2];
    const float* b1    = (const float*)d_in[3];
    const float* Wd3   = (const float*)d_in[4];
    const float* bd3   = (const float*)d_in[5];
    const float* W2d3  = (const float*)d_in[6];
    const float* b2d3  = (const float*)d_in[7];
    const float* Wf    = (const float*)d_in[8];
    const float* bfb   = (const float*)d_in[9];
    const int*   rows  = (const int*)d_in[10];
    const int*   cols  = (const int*)d_in[11];
    const int*   spiral= (const int*)d_in[12];
    float* out = (float*)d_out;

    char* ws = (char*)d_ws;
    float*  poolF = (float*)(ws + OFF_POOLF);
    __bf16* poolH = (__bf16*)(ws + OFF_POOLH);
    __bf16* W1p   = (__bf16*)(ws + OFF_W1P);
    __bf16* Wfp   = (__bf16*)(ws + OFF_WFP);
    __bf16* W2p   = (__bf16*)(ws + OFF_W2P);
    __bf16* Wdp   = (__bf16*)(ws + OFF_WDP);

    // 1) zero pool (16*4096*128 floats, float4-vectorized)
    k_zero_pool<<<8192, 256, 0, stream>>>(poolF);
    // 2) sparse scatter-add pool
    k_pool_scatter<<<NNZ_, C_IN_, 0, stream>>>(x, vals, rows, cols, poolF);
    // 3) fp32 -> bf16
    k_cvt_pool<<<8192, 256, 0, stream>>>(poolF, poolH);
    // 4) pack weights into WMMA-B tiles (64 + 576 + 192 + 96 = 928 tiles)
    k_pack_weights<<<928, 256, 0, stream>>>(W1, Wf, W2d3, Wd3, W1p, Wfp, W2p, Wdp);
    // 5) fused spiral-gather WMMA GEMM + bias + ReLU, 2 vertices per workgroup
    k_spiral_gemm<<<V_OUT_ / 2, 256, 0, stream>>>(poolH, W1p, Wfp, W2p, Wdp,
                                                  b1, bd3, b2d3, bfb, spiral, out);
    (void)in_sizes; (void)n_in; (void)out_size; (void)ws_size;
}
